// QKVAttention_8650064134593
// MI455X (gfx1250) — matrix operand order repaired and checked by reading, not compiled, not run on hardware
//
#include <hip/hip_runtime.h>

// QKV attention (flash-style), MI455X / gfx1250, wave32 WMMA f16->f32,
// TDM (tensor_load_to_lds) double-buffered K/V tile staging.
// qkv: [N=16][3C=192][T=4096] f32, out: [N][C=64][T] f32.
//
// Pass 1: convert K -> f16 [b][t][c] (transposed) and V -> f16 [b][c][t] in d_ws.
// Pass 2: flash attention; K/V tiles are contiguous f16 blocks pulled into LDS
//         by the Tensor Data Mover, overlapped with WMMA compute.

typedef _Float16 v16h __attribute__((ext_vector_type(16)));
typedef _Float16 v8h  __attribute__((ext_vector_type(8)));
typedef _Float16 v4h  __attribute__((ext_vector_type(4)));
typedef float    v8f  __attribute__((ext_vector_type(8)));
typedef unsigned int u32x4 __attribute__((ext_vector_type(4)));
typedef int          i32x4 __attribute__((ext_vector_type(4)));
typedef int          i32x8 __attribute__((ext_vector_type(8)));

#define TN 4096
#define CC 64
#define BM 64   // query rows per block (16 per wave, 4 waves)
#define BS 64   // key/value positions per tile
#define NIT (TN / BS)

// ---------------- pass 1: f32 -> f16 preconversion ----------------
__global__ __launch_bounds__(256)
void preconvert_kernel(const float* __restrict__ qkv,
                       _Float16* __restrict__ Kh,   // [N][T][C]
                       _Float16* __restrict__ Vh) { // [N][C][T]
    size_t i4 = ((size_t)blockIdx.x * 256 + threadIdx.x) * 4;  // index into [N][C][T]
    int b = (int)(i4 >> 18);          // C*T = 2^18
    int c = (int)(i4 >> 12) & 63;
    int t = (int)i4 & 4095;
    const float* kp = qkv + ((size_t)b * 192 + 64  + c) * TN + t;
    const float* vp = qkv + ((size_t)b * 192 + 128 + c) * TN + t;
    float4 k4 = *(const float4*)kp;
    float4 v4 = *(const float4*)vp;
    v4h vh; vh[0] = (_Float16)v4.x; vh[1] = (_Float16)v4.y;
            vh[2] = (_Float16)v4.z; vh[3] = (_Float16)v4.w;
    *(v4h*)&Vh[((size_t)b * CC + c) * TN + t] = vh;
    _Float16* kd = Kh + ((size_t)b * TN + t) * CC + c;   // transpose: [t][c]
    kd[0 * CC] = (_Float16)k4.x;
    kd[1 * CC] = (_Float16)k4.y;
    kd[2 * CC] = (_Float16)k4.z;
    kd[3 * CC] = (_Float16)k4.w;
}

// ---------------- TDM helper: 2D f16 tile -> LDS ----------------
__device__ __forceinline__ unsigned lds_off(const void* p) {
    // gfx1250: generic LDS address = SHARED_BASE (hi 32) | byte offset (lo 32)
    return (unsigned)(unsigned long long)p;
}

__device__ __forceinline__ void tdm_load_tile(const _Float16* gsrc, unsigned ldsaddr,
                                              unsigned tile_d0, unsigned tile_d1,
                                              unsigned long long stride0_elems) {
    unsigned long long ga = (unsigned long long)gsrc;
    const unsigned td0 = 0x40000000u, td1 = 0x40000000u;   // huge -> no OOB clamp
    u32x4 g0;
    g0[0] = 1u;                                            // count=1 (valid user D#)
    g0[1] = ldsaddr;                                       // lds_addr [63:32]
    g0[2] = (unsigned)(ga & 0xFFFFFFFFu);                  // global_addr [95:64]
    g0[3] = (unsigned)((ga >> 32) & 0x01FFFFFFu) | (2u << 30); // ga[56:32] | type=2
    i32x8 g1;
    g1[0] = (int)(1u << 16);                               // data_size=1 (2 bytes)
    g1[1] = (int)((td0 & 0xFFFFu) << 16);                  // tensor_dim0[15:0] @48
    g1[2] = (int)(((td0 >> 16) & 0xFFFFu) | ((td1 & 0xFFFFu) << 16));
    g1[3] = (int)(((td1 >> 16) & 0xFFFFu) | ((tile_d0 & 0xFFFFu) << 16));
    g1[4] = (int)(tile_d1 & 0xFFFFu);                      // tile_dim1; tile_dim2=0
    g1[5] = (int)(unsigned)(stride0_elems & 0xFFFFFFFFu);  // tensor_dim0_stride
    g1[6] = (int)(unsigned)((stride0_elems >> 32) & 0xFFFFu);
    g1[7] = 0;                                             // tensor_dim1_stride=0
    i32x4 z4 = {0, 0, 0, 0};
    i32x8 z8 = {0, 0, 0, 0, 0, 0, 0, 0};
    __builtin_amdgcn_tensor_load_to_lds(g0, g1, z4, z4, z8, 0);
}

// ---------------- pass 2: flash attention ----------------
__global__ __launch_bounds__(128)
void qkv_attn_wmma_kernel(const float* __restrict__ qkv,
                          const _Float16* __restrict__ Kh,
                          const _Float16* __restrict__ Vh,
                          float* __restrict__ out) {
    const int b    = blockIdx.y;
    const int t0   = blockIdx.x * BM;
    const int tid  = threadIdx.x;
    const int w    = tid >> 5;        // wave id 0..3
    const int lane = tid & 31;
    const int ln   = lane & 15;
    const int hi   = lane >> 4;       // 0 or 1 (wave half)

    const float*    qb  = qkv + (size_t)b * (3 * CC) * TN;     // q: [c][t] f32
    const _Float16* khb = Kh + (size_t)b * TN * CC;            // [t][c] f16
    const _Float16* vhb = Vh + (size_t)b * CC * TN;            // [c][t] f16
    float*          ob  = out + (size_t)b * CC * TN;

    __shared__ _Float16 Ks[2][BS][CC];    // [s][c], 8 KB x2 (TDM: contiguous block)
    __shared__ _Float16 Vs[2][CC][BS];    // [c][s], 8 KB x2 (TDM: 2D 64x64, stride T)
    __shared__ _Float16 Ps[4][16][BS];    // per-wave P bounce [t][s], 8 KB

    // ---- resident Q A-fragments (16x32 f16 each), pre-scaled by 1/sqrt(C) ----
    const int trow = t0 + w * 16;
    const int tq   = trow + ln;                 // M = ln
    v16h qfrag[2];
    #pragma unroll
    for (int ct = 0; ct < 2; ++ct) {
        #pragma unroll
        for (int e = 0; e < 16; ++e) {
            int c = ct * 32 + hi * 16 + e;
            qfrag[ct][e] = (_Float16)(qb[(size_t)c * TN + tq] * 0.125f);
        }
    }

    v8f acc[4];
    #pragma unroll
    for (int ct = 0; ct < 4; ++ct)
        #pragma unroll
        for (int g = 0; g < 8; ++g) acc[ct][g] = 0.0f;

    float mrow[8], lrow[8];
    #pragma unroll
    for (int g = 0; g < 8; ++g) { mrow[g] = -1e30f; lrow[g] = 0.0f; }

    // ---- prologue: start DMA of tile 0 into buffer 0 ----
    if (w == 0) {
        tdm_load_tile(khb, lds_off(&Ks[0][0][0]), BS * CC, 0, BS * CC);
        tdm_load_tile(vhb, lds_off(&Vs[0][0][0]), BS, CC, TN);
    }

    int buf = 0;
    for (int it = 0; it < NIT; ++it) {
        const int s0 = it * BS;
        __syncthreads();   // all waves done reading buf^1 (previous tile)
        if (w == 0) {
            if (it + 1 < NIT) {
                const int sn = s0 + BS;
                tdm_load_tile(khb + (size_t)sn * CC, lds_off(&Ks[buf ^ 1][0][0]),
                              BS * CC, 0, BS * CC);
                tdm_load_tile(vhb + sn, lds_off(&Vs[buf ^ 1][0][0]), BS, CC, TN);
                __builtin_amdgcn_s_wait_tensorcnt(2);  // retire current pair, next in flight
            } else {
                __builtin_amdgcn_s_wait_tensorcnt(0);
            }
        }
        __syncthreads();   // current tile published to all waves

        // ---- W = (Q^T)(K): four 16x16 s-tiles, K-dim = 64 = 2 chunks of 32 ----
        v8f wt[4];
        #pragma unroll
        for (int st = 0; st < 4; ++st) {
            v8f d;
            #pragma unroll
            for (int g = 0; g < 8; ++g) d[g] = 0.0f;
            #pragma unroll
            for (int ct = 0; ct < 2; ++ct) {
                // B frag: N = s = st*16+ln ; K = c = ct*32 + hi*16 + e
                v16h bfrag = *(const v16h*)&Ks[buf][st * 16 + ln][ct * 32 + hi * 16];
                d = __builtin_amdgcn_wmma_f32_16x16x32_f16(
                        false, qfrag[ct], false, bfrag, (short)0, d, false, false);
            }
            wt[st] = d;
        }

        // ---- online softmax (lane half hi holds rows 8*hi+g, each reduced over its 16 lanes) ----
        float alpha[8];
        #pragma unroll
        for (int g = 0; g < 8; ++g) {
            float mx = fmaxf(fmaxf(wt[0][g], wt[1][g]), fmaxf(wt[2][g], wt[3][g]));
            mx = fmaxf(mx, __shfl_xor(mx, 1));
            mx = fmaxf(mx, __shfl_xor(mx, 2));
            mx = fmaxf(mx, __shfl_xor(mx, 4));
            mx = fmaxf(mx, __shfl_xor(mx, 8));
            float mnew = fmaxf(mrow[g], mx);
            float a = __expf(mrow[g] - mnew);
            float p0 = __expf(wt[0][g] - mnew);
            float p1 = __expf(wt[1][g] - mnew);
            float p2 = __expf(wt[2][g] - mnew);
            float p3 = __expf(wt[3][g] - mnew);
            float rs = (p0 + p1) + (p2 + p3);
            rs += __shfl_xor(rs, 1);
            rs += __shfl_xor(rs, 2);
            rs += __shfl_xor(rs, 4);
            rs += __shfl_xor(rs, 8);
            lrow[g]  = lrow[g] * a + rs;
            mrow[g]  = mnew;
            alpha[g] = a;
            Ps[w][g + hi * 8][ln]      = (_Float16)p0;   // [row = g+8*hi][s]
            Ps[w][g + hi * 8][16 + ln] = (_Float16)p1;
            Ps[w][g + hi * 8][32 + ln] = (_Float16)p2;
            Ps[w][g + hi * 8][48 + ln] = (_Float16)p3;
        }

        #pragma unroll
        for (int ct = 0; ct < 4; ++ct)
            #pragma unroll
            for (int g = 0; g < 8; ++g) acc[ct][g] *= alpha[g];

        // ---- reload P as two A-fragments (k-chunks of 32) ----
        v16h pfrag[2];
        #pragma unroll
        for (int ck = 0; ck < 2; ++ck) {
            v8h plo = *(const v8h*)&Ps[w][ln][ck * 32 + hi * 16];
            v8h phi = *(const v8h*)&Ps[w][ln][ck * 32 + hi * 16 + 8];
            pfrag[ck] = __builtin_shufflevector(plo, phi,
                            0, 1, 2, 3, 4, 5, 6, 7, 8, 9, 10, 11, 12, 13, 14, 15);
        }

        // ---- O^T += P (V^T): 4 c-chunks x 2 k-chunks ----
        #pragma unroll
        for (int ck = 0; ck < 2; ++ck)
            #pragma unroll
            for (int ct = 0; ct < 4; ++ct) {
                // B frag: N = c = ct*16+ln ; K = s = ck*32 + hi*16 + e
                v16h vfrag = *(const v16h*)&Vs[buf][ct * 16 + ln][ck * 32 + hi * 16];
                acc[ct] = __builtin_amdgcn_wmma_f32_16x16x32_f16(
                              false, pfrag[ck], false, vfrag, (short)0, acc[ct], false, false);
            }

        buf ^= 1;
    }

    // ---- epilogue ----
    float inv[8];
    #pragma unroll
    for (int g = 0; g < 8; ++g) inv[g] = 1.0f / lrow[g];
    #pragma unroll
    for (int ct = 0; ct < 4; ++ct) {
        int c = ct * 16 + ln;
        float* op = ob + (size_t)c * TN + trow + hi * 8;
        #pragma unroll
        for (int g = 0; g < 8; ++g) op[g] = acc[ct][g] * inv[g];
    }
}

extern "C" void kernel_launch(void* const* d_in, const int* in_sizes, int n_in,
                              void* d_out, int out_size, void* d_ws, size_t ws_size,
                              hipStream_t stream) {
    const float* qkv = (const float*)d_in[0];
    float* out = (float*)d_out;
    _Float16* Kh = (_Float16*)d_ws;                      // 8 MB
    _Float16* Vh = Kh + (size_t)16 * TN * CC;            // 8 MB
    preconvert_kernel<<<4096, 256, 0, stream>>>(qkv, Kh, Vh);
    qkv_attn_wmma_kernel<<<dim3(TN / BM, 16), 128, 0, stream>>>(qkv, Kh, Vh, out);
}
